// Linear_66331474920136
// MI455X (gfx1250) — matrix operand order repaired and checked by reading, not compiled, not run on hardware
//
#include <hip/hip_runtime.h>

#define NUM_TOK   8192           // 4 * 2048
#define IN_DIM    1024
#define OUT_DIM   1024
#define NEXP      8
#define CAP       16384          // worst-case assignments per expert
#define PLANE_ELEMS ((size_t)NUM_TOK * IN_DIM)   // == NEXP*OUT_DIM*IN_DIM == 8388608

// fast (pre-split) path: 128-token tiles
#define MT_PRE    128
#define MAXTILES_PRE  (NUM_TOK*2/MT_PRE + NEXP)  // 136
#define KCP       64
#define LDAP      72             // padded LDS row stride (ushorts)

// fallback (on-the-fly) path: 16-token tiles
#define MAXTILES_FLY  (NUM_TOK*2/16 + NEXP)      // 1032
#define KCF       128
#define LDAF      136

typedef unsigned short ushort_t;
typedef __attribute__((ext_vector_type(16))) __bf16 v16bf;
typedef __attribute__((ext_vector_type(8)))  float  v8f;
typedef int v4i_t __attribute__((vector_size(16)));

union V16 { v16bf v; unsigned short u[16]; uint4 q[2]; };

__device__ __forceinline__ unsigned short f2bf(float f) {
    unsigned u = __builtin_bit_cast(unsigned, f);
    return (unsigned short)((u + 0x7FFFu + ((u >> 16) & 1u)) >> 16);   // RNE
}
__device__ __forceinline__ float bf2f(unsigned short h) {
    return __builtin_bit_cast(float, ((unsigned)h) << 16);
}

// ---- gfx1250 async global->LDS copy (ASYNCcnt-tracked), with sync fallback ---
#if defined(__has_builtin)
#if __has_builtin(__builtin_amdgcn_global_load_async_to_lds_b128)
#define HAVE_ASYNC_LDS 1
#endif
#endif

__device__ __forceinline__ void cp16_g2lds(void* lds, const void* g) {
#ifdef HAVE_ASYNC_LDS
    __builtin_amdgcn_global_load_async_to_lds_b128(
        (__attribute__((address_space(1))) v4i_t*)(g),
        (__attribute__((address_space(3))) v4i_t*)(lds), 0, 0);
#else
    *(uint4*)lds = *(const uint4*)g;
#endif
}
__device__ __forceinline__ void wait_async_copies() {
#ifdef HAVE_ASYNC_LDS
#if __has_builtin(__builtin_amdgcn_s_wait_asynccnt)
    __builtin_amdgcn_s_wait_asynccnt(0);
#else
    asm volatile("s_wait_asynccnt 0x0" ::: "memory");
#endif
#endif
}

// ---------------- split f32 -> (hi,lo) bf16 planes ----------------------------
__global__ __launch_bounds__(256) void split_planes(
    const float4* __restrict__ src, ushort_t* __restrict__ hi, ushort_t* __restrict__ lo)
{
    const int i = blockIdx.x * 256 + threadIdx.x;           // over PLANE_ELEMS/4
    const float4 v = src[i];
    const unsigned short h0 = f2bf(v.x), h1 = f2bf(v.y), h2 = f2bf(v.z), h3 = f2bf(v.w);
    const unsigned short l0 = f2bf(v.x - bf2f(h0)), l1 = f2bf(v.y - bf2f(h1));
    const unsigned short l2 = f2bf(v.z - bf2f(h2)), l3 = f2bf(v.w - bf2f(h3));
    *(uint2*)(hi + (size_t)i * 4) = make_uint2((unsigned)h0 | ((unsigned)h1 << 16),
                                               (unsigned)h2 | ((unsigned)h3 << 16));
    *(uint2*)(lo + (size_t)i * 4) = make_uint2((unsigned)l0 | ((unsigned)l1 << 16),
                                               (unsigned)l2 | ((unsigned)l3 << 16));
}

// ---------------- zero routing metadata ---------------------------------------
__global__ void zero_meta(int* counts, int* ntiles) {
    if (threadIdx.x < NEXP) counts[threadIdx.x] = 0;
    if (threadIdx.x == NEXP) *ntiles = 0;
}

// ---------------- gate logits, top-2, routing lists ---------------------------
__global__ __launch_bounds__(256) void gate_topk(
    const float* __restrict__ x, const float* __restrict__ Wg, const float* __restrict__ bg,
    int* __restrict__ counts, int* __restrict__ lists,
    int* __restrict__ e0a, int* __restrict__ e1a,
    float* __restrict__ w0a, float* __restrict__ w1a)
{
    const int t    = blockIdx.x * 8 + (threadIdx.x >> 5);  // one wave32 per token
    const int lane = threadIdx.x & 31;
    const float* xr = x + ((size_t)t << 10);

    float acc[NEXP] = {0.f,0.f,0.f,0.f,0.f,0.f,0.f,0.f};
    for (int i = lane; i < IN_DIM; i += 32) {
        const float xv = xr[i];
        const float* wr = Wg + i * NEXP;
        #pragma unroll
        for (int e = 0; e < NEXP; ++e) acc[e] += xv * wr[e];
    }
    #pragma unroll
    for (int off = 16; off > 0; off >>= 1) {
        #pragma unroll
        for (int e = 0; e < NEXP; ++e) acc[e] += __shfl_xor(acc[e], off, 32);
    }
    if (lane == 0) {
        #pragma unroll
        for (int e = 0; e < NEXP; ++e) acc[e] += bg[e];
        int i0 = 0; float l0 = acc[0];
        #pragma unroll
        for (int e = 1; e < NEXP; ++e) if (acc[e] > l0) { l0 = acc[e]; i0 = e; }
        int i1 = (i0 == 0) ? 1 : 0; float l1 = acc[i1];
        #pragma unroll
        for (int e = 0; e < NEXP; ++e) if (e != i0 && acc[e] > l1) { l1 = acc[e]; i1 = e; }
        // renormalized top-2 softmax == sigmoid of logit gap
        const float g0 = 1.0f / (1.0f + __expf(l1 - l0));
        const float g1 = 1.0f - g0;
        e0a[t] = i0; e1a[t] = i1; w0a[t] = g0; w1a[t] = g1;
        int p0 = atomicAdd(&counts[i0], 1); lists[i0 * CAP + p0] = (t << 1);
        int p1 = atomicAdd(&counts[i1], 1); lists[i1 * CAP + p1] = (t << 1) | 1;
    }
}

// ---------------- build tile descriptors (runtime tile size) ------------------
__global__ void build_tiles(const int* __restrict__ counts, int4* __restrict__ tiles,
                            int* __restrict__ ntiles, int ts)
{
    __shared__ int base[NEXP], cnt[NEXP];
    if (threadIdx.x == 0) {
        int s = 0;
        for (int e = 0; e < NEXP; ++e) { cnt[e] = counts[e]; base[e] = s; s += (cnt[e] + ts - 1) / ts; }
        *ntiles = s;
    }
    __syncthreads();
    for (int e = 0; e < NEXP; ++e) {
        const int c = cnt[e], nt = (c + ts - 1) / ts;
        for (int i = threadIdx.x; i < nt; i += blockDim.x)
            tiles[base[e] + i] = make_int4(e, i * ts, min(ts, c - i * ts), 0);
    }
}

// ---------------- out = w0*be[e0] + w1*be[e1] ---------------------------------
__global__ __launch_bounds__(256) void bias_init(
    const float* __restrict__ be, const int* __restrict__ e0, const int* __restrict__ e1,
    const float* __restrict__ w0, const float* __restrict__ w1, float* __restrict__ out)
{
    const int idx = blockIdx.x * 256 + threadIdx.x;
    const int t = idx >> 10, o = idx & (OUT_DIM - 1);
    out[idx] = w0[t] * be[(e0[t] << 10) + o] + w1[t] * be[(e1[t] << 10) + o];
}

// =====================================================================
// FAST PATH: routed GEMM on pre-split hi/lo bf16 planes.
// Block: 256 thr = 8 waves. Tile: 128 gathered tokens x 128 cols.
// Each wave owns 16 cols and 8 M-subtiles -> B operand reused 8x.
// A tile staged via async global->LDS copies (ASYNCcnt path on gfx1250).
// =====================================================================
__global__ __launch_bounds__(256) void moe_gemm_pre(
    const ushort_t* __restrict__ hiX, const ushort_t* __restrict__ loX,
    const ushort_t* __restrict__ hiW, const ushort_t* __restrict__ loW,
    float* __restrict__ out,
    const int4* __restrict__ tiles, const int* __restrict__ ntiles,
    const int* __restrict__ lists, const float* __restrict__ w0, const float* __restrict__ w1)
{
    const int tile = blockIdx.x;
    if (tile >= *ntiles) return;
    const int4 td = tiles[tile];
    const int e = td.x, start = td.y, cnt = td.z;
    const int n0 = blockIdx.y * 128;

    __shared__ __align__(16) unsigned short AsH[MT_PRE][LDAP];
    __shared__ __align__(16) unsigned short AsL[MT_PRE][LDAP];
    __shared__ int   toks[MT_PRE];
    __shared__ float wts[MT_PRE];

    const int tid  = threadIdx.x;
    const int lane = tid & 31;
    const int wave = tid >> 5;

    for (int i = tid; i < MT_PRE; i += 256) {
        const int entry = (i < cnt) ? lists[e * CAP + start + i] : -1;
        const int t = (entry >= 0) ? (entry >> 1) : 0;
        toks[i] = t;
        wts[i]  = (entry >= 0) ? ((entry & 1) ? w1[t] : w0[t]) : 0.0f;  // pad rows weight 0
    }
    __syncthreads();

    const int col = lane & 15;
    const int n   = n0 + wave * 16 + col;
    const size_t wbase = ((size_t)e << 20) + ((size_t)n << 10);
    const int kb = (lane < 16) ? 0 : 16;   // B operand: lanes 16-31 hold K=16..31
    const int ka = (lane < 16) ? 0 : 16;   // A operand: lanes 16-31 hold K=16..31
    const int m  = lane & 15;

    // staging addresses for this thread (4 chunks of 16B per plane per k-step)
    const int sr = tid >> 3;               // base row handled by this thread
    const int sc = (tid & 7) << 3;         // element column (x8)

    v8f acc[8] = {};

    for (int k0 = 0; k0 < IN_DIM; k0 += KCP) {
        // stage A planes: 128 rows x 64 k, async global->LDS, 16B granules
        #pragma unroll
        for (int it = 0; it < 4; ++it) {
            const int r = sr + it * 32;
            const size_t g = ((size_t)toks[r] << 10) + k0 + sc;
            cp16_g2lds(&AsH[r][sc], hiX + g);
            cp16_g2lds(&AsL[r][sc], loX + g);
        }
        wait_async_copies();
        __syncthreads();

        // prefetch next K-chunk of this lane's B stream
        if (k0 + KCP < IN_DIM) {
            __builtin_prefetch((const void*)(hiW + wbase + k0 + KCP + kb), 0, 0);
            __builtin_prefetch((const void*)(loW + wbase + k0 + KCP + kb), 0, 0);
        }

        #pragma unroll
        for (int kk = 0; kk < KCP; kk += 32) {
            V16 bH, bL;
            const ushort_t* bh = hiW + wbase + k0 + kk + kb;
            const ushort_t* bl = loW + wbase + k0 + kk + kb;
            bH.q[0] = *(const uint4*)(bh);     bH.q[1] = *(const uint4*)(bh + 8);
            bL.q[0] = *(const uint4*)(bl);     bL.q[1] = *(const uint4*)(bl + 8);

            #pragma unroll
            for (int s = 0; s < 8; ++s) {
                V16 aH, aL;
                const int mr = s * 16 + m;
                aH.q[0] = *(const uint4*)&AsH[mr][kk + ka];
                aH.q[1] = *(const uint4*)&AsH[mr][kk + ka + 8];
                aL.q[0] = *(const uint4*)&AsL[mr][kk + ka];
                aL.q[1] = *(const uint4*)&AsL[mr][kk + ka + 8];
                // split-bf16: aH*bH + aH*bL + aL*bH  (~fp32 accuracy)
                acc[s] = __builtin_amdgcn_wmma_f32_16x16x32_bf16(false, aL.v, false, bH.v, (short)0, acc[s], false, false);
                acc[s] = __builtin_amdgcn_wmma_f32_16x16x32_bf16(false, aH.v, false, bL.v, (short)0, acc[s], false, false);
                acc[s] = __builtin_amdgcn_wmma_f32_16x16x32_bf16(false, aH.v, false, bH.v, (short)0, acc[s], false, false);
            }
        }
        __syncthreads();
    }

    const int mbase = (lane < 16) ? 0 : 8;
    #pragma unroll
    for (int s = 0; s < 8; ++s) {
        #pragma unroll
        for (int j = 0; j < 8; ++j) {
            const int mm = s * 16 + mbase + j;
            atomicAdd(out + ((size_t)toks[mm] << 10) + n, wts[mm] * acc[s][j]);
        }
    }
}

// =====================================================================
// FALLBACK PATH (small workspace): on-the-fly split, 16-token tiles
// =====================================================================
__global__ __launch_bounds__(256) void moe_gemm_fly(
    const float* __restrict__ x, const float* __restrict__ We, float* __restrict__ out,
    const int4* __restrict__ tiles, const int* __restrict__ ntiles,
    const int* __restrict__ lists, const float* __restrict__ w0, const float* __restrict__ w1)
{
    const int tile = blockIdx.x;
    if (tile >= *ntiles) return;
    const int4 td = tiles[tile];
    const int e = td.x, start = td.y, cnt = td.z;
    const int n0 = blockIdx.y * 128;

    __shared__ __align__(16) unsigned short AsH[16][LDAF];
    __shared__ __align__(16) unsigned short AsL[16][LDAF];
    __shared__ int   toks[16];
    __shared__ float wts[16];

    const int tid  = threadIdx.x;
    const int lane = tid & 31;
    const int wave = tid >> 5;

    if (tid < 16) {
        const int entry = (tid < cnt) ? lists[e * CAP + start + tid] : -1;
        const int t = (entry >= 0) ? (entry >> 1) : 0;
        toks[tid] = t;
        wts[tid]  = (entry >= 0) ? ((entry & 1) ? w1[t] : w0[t]) : 0.0f;
    }
    __syncthreads();

    const int col = lane & 15;
    const int n   = n0 + wave * 16 + col;
    const float* __restrict__ Wrow = We + ((size_t)e << 20) + ((size_t)n << 10);
    const int kb = (lane < 16) ? 0 : 16;
    const int ka = (lane < 16) ? 0 : 16;
    const int m  = lane & 15;

    v8f acc = {};

    for (int k0 = 0; k0 < IN_DIM; k0 += KCF) {
        #pragma unroll
        for (int it = 0; it < 2; ++it) {
            const int i = tid + it * 256;
            const int r = i >> 5;
            const int c = (i & 31) << 2;
            const float4 v = *(const float4*)(x + ((size_t)toks[r] << 10) + k0 + c);
            const unsigned short h0 = f2bf(v.x), h1 = f2bf(v.y), h2 = f2bf(v.z), h3 = f2bf(v.w);
            const unsigned short l0 = f2bf(v.x - bf2f(h0)), l1 = f2bf(v.y - bf2f(h1));
            const unsigned short l2 = f2bf(v.z - bf2f(h2)), l3 = f2bf(v.w - bf2f(h3));
            *(uint2*)&AsH[r][c] = make_uint2((unsigned)h0 | ((unsigned)h1 << 16),
                                             (unsigned)h2 | ((unsigned)h3 << 16));
            *(uint2*)&AsL[r][c] = make_uint2((unsigned)l0 | ((unsigned)l1 << 16),
                                             (unsigned)l2 | ((unsigned)l3 << 16));
        }
        __syncthreads();

        #pragma unroll
        for (int kk = 0; kk < KCF; kk += 32) {
            V16 aH, aL, bH, bL;
            aH.q[0] = *(const uint4*)&AsH[m][kk + ka];
            aH.q[1] = *(const uint4*)&AsH[m][kk + ka + 8];
            aL.q[0] = *(const uint4*)&AsL[m][kk + ka];
            aL.q[1] = *(const uint4*)&AsL[m][kk + ka + 8];

            const float* bp = Wrow + k0 + kk + kb;
            #pragma unroll
            for (int q = 0; q < 4; ++q) {
                const float4 f = *(const float4*)(bp + q * 4);
                unsigned short h;
                h = f2bf(f.x); bH.u[q*4+0] = h; bL.u[q*4+0] = f2bf(f.x - bf2f(h));
                h = f2bf(f.y); bH.u[q*4+1] = h; bL.u[q*4+1] = f2bf(f.y - bf2f(h));
                h = f2bf(f.z); bH.u[q*4+2] = h; bL.u[q*4+2] = f2bf(f.z - bf2f(h));
                h = f2bf(f.w); bH.u[q*4+3] = h; bL.u[q*4+3] = f2bf(f.w - bf2f(h));
            }
            acc = __builtin_amdgcn_wmma_f32_16x16x32_bf16(false, aL.v, false, bH.v, (short)0, acc, false, false);
            acc = __builtin_amdgcn_wmma_f32_16x16x32_bf16(false, aH.v, false, bL.v, (short)0, acc, false, false);
            acc = __builtin_amdgcn_wmma_f32_16x16x32_bf16(false, aH.v, false, bH.v, (short)0, acc, false, false);
        }
        __syncthreads();
    }

    const int mbase = (lane < 16) ? 0 : 8;
    #pragma unroll
    for (int j = 0; j < 8; ++j) {
        const int mm = mbase + j;
        atomicAdd(out + ((size_t)toks[mm] << 10) + n, wts[mm] * acc[j]);
    }
}

// ------------------------------- launcher -------------------------------------
extern "C" void kernel_launch(void* const* d_in, const int* in_sizes, int n_in,
                              void* d_out, int out_size, void* d_ws, size_t ws_size,
                              hipStream_t stream)
{
    (void)in_sizes; (void)n_in; (void)out_size;
    const float* x  = (const float*)d_in[0];
    const float* Wg = (const float*)d_in[1];
    const float* bg = (const float*)d_in[2];
    const float* We = (const float*)d_in[3];
    const float* be = (const float*)d_in[4];
    float* out = (float*)d_out;

    char* ws = (char*)d_ws;
    int*   counts = (int*)(ws);                       // 8 ints
    int*   ntiles = (int*)(ws + 64);                  // 1 int
    int*   e0     = (int*)(ws + 256);                 // NUM_TOK
    int*   e1     = e0 + NUM_TOK;
    float* w0     = (float*)(e1 + NUM_TOK);
    float* w1     = w0 + NUM_TOK;
    int*   lists  = (int*)(w1 + NUM_TOK);             // NEXP * CAP
    int4*  tiles  = (int4*)(lists + NEXP * CAP);      // MAXTILES_FLY max (16B aligned)

    const size_t rout_end = (size_t)(((char*)(tiles + MAXTILES_FLY)) - ws);
    const size_t plane_off = (rout_end + 255) & ~(size_t)255;
    const size_t plane_bytes = PLANE_ELEMS * sizeof(unsigned short);  // 16 MiB
    const size_t need = plane_off + 4 * plane_bytes;                  // ~64.7 MiB

    // common routing phase
    zero_meta<<<1, 32, 0, stream>>>(counts, ntiles);
    gate_topk<<<NUM_TOK / 8, 256, 0, stream>>>(x, Wg, bg, counts, lists, e0, e1, w0, w1);
    bias_init<<<(NUM_TOK * OUT_DIM) / 256, 256, 0, stream>>>(be, e0, e1, w0, w1, out);

    if (ws_size >= need) {
        ushort_t* hiX = (ushort_t*)(ws + plane_off);
        ushort_t* loX = hiX + PLANE_ELEMS;
        ushort_t* hiW = loX + PLANE_ELEMS;
        ushort_t* loW = hiW + PLANE_ELEMS;
        split_planes<<<(unsigned)(PLANE_ELEMS / 4 / 256), 256, 0, stream>>>((const float4*)x,  hiX, loX);
        split_planes<<<(unsigned)(PLANE_ELEMS / 4 / 256), 256, 0, stream>>>((const float4*)We, hiW, loW);
        build_tiles<<<1, 256, 0, stream>>>(counts, tiles, ntiles, MT_PRE);
        moe_gemm_pre<<<dim3(MAXTILES_PRE, OUT_DIM / 128), 256, 0, stream>>>(
            hiX, loX, hiW, loW, out, tiles, ntiles, lists, w0, w1);
    } else {
        build_tiles<<<1, 256, 0, stream>>>(counts, tiles, ntiles, 16);
        moe_gemm_fly<<<dim3(MAXTILES_FLY, OUT_DIM / 128), 256, 0, stream>>>(
            x, We, out, tiles, ntiles, lists, w0, w1);
    }
}
